// DeepRBFNetwork_6150393168265
// MI455X (gfx1250) — matrix operand order repaired and checked by reading, not compiled, not run on hardware
//
#include <hip/hip_runtime.h>
#include <hip/hip_bf16.h>
#include <math.h>

typedef __attribute__((ext_vector_type(16))) _Float16 v16h;
typedef __attribute__((ext_vector_type(8)))  float    v8f;
typedef __attribute__((ext_vector_type(4)))  unsigned int v4u;
typedef __attribute__((ext_vector_type(4)))  unsigned int v4s;   // SGPR quad
typedef __attribute__((ext_vector_type(8)))  unsigned int v8s;   // SGPR octet

#define B_ROWS 4096
#define DDIM   512
#define KCLS   100
#define MTILE  128          // feature rows per block (4 row-groups x 32)
#define ESTAGE 64           // A_k e-rows staged per LDS buffer (2 e-groups x 32)

#define LDSF_HALVES (MTILE * DDIM)          // 65536 halves = 128 KB
#define ABUF_HALVES (ESTAGE * DDIM)         // 32768 halves = 64 KB
#define LDSF_BYTES  (LDSF_HALVES * 2)       // 131072
#define ABUF_BYTES  (ABUF_HALVES * 2)       // 65536
#define LDS_TOTAL   (LDSF_BYTES + 2 * ABUF_BYTES)  // 262144 B

union HFrag { v16h h; v4u u[2]; };

// ---------------------------------------------------------------------------
// Pre-pass: fp32 -> fp16, 8 elements per thread.
// ---------------------------------------------------------------------------
__global__ void cvt_f32_to_f16(const float* __restrict__ src,
                               _Float16* __restrict__ dst, int n8) {
  int i = blockIdx.x * blockDim.x + threadIdx.x;
  if (i >= n8) return;
  const float4* s = (const float4*)src;
  float4 a = s[2 * i];
  float4 b = s[2 * i + 1];
  _Float16 h[8] = {(_Float16)a.x, (_Float16)a.y, (_Float16)a.z, (_Float16)a.w,
                   (_Float16)b.x, (_Float16)b.y, (_Float16)b.z, (_Float16)b.w};
  ((v4u*)dst)[i] = *(v4u*)h;
}

// ---------------------------------------------------------------------------
// Async global->LDS staging (GLOBAL_LOAD_ASYNC_TO_LDS_B128, ASYNCcnt).
// ---------------------------------------------------------------------------
__device__ __forceinline__ void stage_async(unsigned ldsBase,
                                            const _Float16* src, int n16) {
  unsigned long long g = (unsigned long long)(uintptr_t)src;  // block-uniform
  unsigned off = threadIdx.x * 16u;
#pragma unroll
  for (int i = 0; i < n16; ++i) {
    unsigned l = ldsBase + off;
    asm volatile("global_load_async_to_lds_b128 %0, %1, %2"
                 :: "v"(l), "v"(off), "s"(g)
                 : "memory");
    off += 4096u;  // 256 threads * 16 B
  }
}

__device__ __forceinline__ void wait_async0() {
#if __has_builtin(__builtin_amdgcn_s_wait_asynccnt)
  __builtin_amdgcn_s_wait_asynccnt(0);
#else
  asm volatile("s_wait_asynccnt 0x0" ::: "memory");
#endif
}

__device__ __forceinline__ void wait_tensor0() {
#if __has_builtin(__builtin_amdgcn_s_wait_tensorcnt)
  __builtin_amdgcn_s_wait_tensorcnt(0);
#else
  asm volatile("s_wait_tensorcnt 0x0" ::: "memory");
#endif
}

// ---------------------------------------------------------------------------
// Tensor Data Mover: one-issue DMA of a 64x512 f16 tile (64 KB) into LDS.
// D# group 0: count=1 | lds_addr | global_addr[56:0] | type=2.
// D# group 1 (compile-time const): data_size=2B, tensor_dim0=512,
// tensor_dim1=64, tile_dim0=512, tile_dim1=64, tensor_dim0_stride=512.
// Issued by one wave only; tracked on TENSORcnt.
// ---------------------------------------------------------------------------
__device__ __forceinline__ void tdm_load_tile(unsigned ldsByteOff,
                                              const _Float16* src) {
  unsigned long long ga = (unsigned long long)(uintptr_t)src;
  v4s g0;
  g0[0] = 1u;                                               // count=1 (valid)
  g0[1] = ldsByteOff;                                       // lds_addr
  g0[2] = (unsigned)ga;                                     // global_addr[31:0]
  g0[3] = (unsigned)((ga >> 32) & 0x01FFFFFFu) | (2u << 30); // ga[56:32]|type=2
  v8s g1;
  g1[0] = 0x00010000u;               // workgroup_mask=0, data_size=1 (2 bytes)
  g1[1] = (unsigned)DDIM << 16;      // tensor_dim0[15:0] = 512  (bits 79:48)
  g1[2] = (unsigned)ESTAGE << 16;    // tensor_dim1[15:0] = 64   (bits 111:80)
  g1[3] = (unsigned)DDIM << 16;      // tile_dim0 = 512          (bits 127:112)
  g1[4] = (unsigned)ESTAGE;          // tile_dim1 = 64, tile_dim2 = 0
  g1[5] = (unsigned)DDIM;            // tensor_dim0_stride = 512 (bits 207:160)
  g1[6] = 0u;                        // stride hi / tensor_dim1_stride lo
  g1[7] = 0u;                        // tensor_dim1_stride hi
  asm volatile("tensor_load_to_lds %0, %1" :: "s"(g0), "s"(g1) : "memory");
}

// ---------------------------------------------------------------------------
// Fused GEMM + row-norm with 32x32 register macro-tiles per wave:
//   out[b,k] = sqrt( sum_e ( sum_d feat[b,d]*A[k,e,d] + bias[k,e] )^2 )
// WMMA-A = A_k tile (M=e, K=d), WMMA-B = features (K=d, N=b row).
// 8 waves = 4 row-groups x 2 e-groups; 4 WMMA per d-step from 2+2 fragments.
// A e-tiles staged by TDM (TENSORcnt); feature tile by async-to-LDS (ASYNCcnt).
// ---------------------------------------------------------------------------
__global__ __launch_bounds__(256, 1)
void rbf_wmma_kernel(const _Float16* __restrict__ featH,  // [4096][512] f16
                     const _Float16* __restrict__ aH,     // [100][512][512] f16 (e,d)
                     const float*    __restrict__ bias,   // [100][512] f32
                     float* __restrict__ out)             // [4096][100] f32
{
  extern __shared__ _Float16 lds[];
  _Float16* ldsF = lds;                    // feature tile
  _Float16* ldsA = lds + LDSF_HALVES;      // double-buffered A e-tiles

  const int tid    = threadIdx.x;
  const int lane   = tid & 31;
  const int wave   = tid >> 5;
  const int rowGrp = wave & 3;             // which 32 feature rows
  const int eGrp   = wave >> 2;            // which 32 e's of the staged 64
  const int m0     = blockIdx.x * MTILE;
  const int k      = blockIdx.y;
  const int hi     = lane >> 4;            // lane half (K sub-range select)

  const _Float16* aCls = aH + (size_t)k * DDIM * DDIM;

  // Stage feature tile (async-to-LDS) and A e-tile 0 (TDM, wave 0 only).
  stage_async(0u, featH + (size_t)m0 * DDIM, LDSF_BYTES / 4096);
  if (wave == 0) tdm_load_tile((unsigned)LDSF_BYTES, aCls);
  wait_async0();
  if (wave == 0) wait_tensor0();
  __syncthreads();

  // Fragment base pointers (ISA 7.12.2 16-bit layouts).
  const _Float16* aBase0 = ldsA + (size_t)(eGrp * 32 + (lane & 15)) * DDIM + (hi << 3);
  const _Float16* fBase  = ldsF + (size_t)(rowGrp * 32 + (lane & 15)) * DDIM + (hi << 3);

  float accp0 = 0.f, accp1 = 0.f;   // row-subtile 0 / 1 partial norms

  for (int et = 0; et < DDIM / ESTAGE; ++et) {
    const int buf = et & 1;
    // TDM-stage next e-tile into the other buffer (overlaps 64 WMMAs below).
    if (wave == 0 && et + 1 < DDIM / ESTAGE) {
      tdm_load_tile((unsigned)(LDSF_BYTES + (buf ^ 1) * ABUF_BYTES),
                    aCls + (size_t)(et + 1) * ESTAGE * DDIM);
    }

    const _Float16* aB = aBase0 + (size_t)buf * ABUF_HALVES;

    // Bias for this lane's accumulator rows: e = et*64 + eGrp*32 + s*16 + hi*8 + r
    const float* bp = bias + (size_t)k * DDIM + et * ESTAGE + eGrp * 32 + (hi << 3);
    float4 be0a = ((const float4*)bp)[0];
    float4 be0b = ((const float4*)bp)[1];
    float4 be1a = ((const float4*)(bp + 16))[0];
    float4 be1b = ((const float4*)(bp + 16))[1];

    v8f c00 = {}, c01 = {}, c10 = {}, c11 = {};
#pragma unroll
    for (int d0 = 0; d0 < DDIM; d0 += 32) {
      HFrag a0, a1, b0, b1;
      a0.u[0] = *(const v4u*)(aB + d0);
      a0.u[1] = *(const v4u*)(aB + d0 + 16);
      a1.u[0] = *(const v4u*)(aB + 8192 + d0);          // e-subtile 1 (+16 rows)
      a1.u[1] = *(const v4u*)(aB + 8192 + d0 + 16);
      b0.u[0] = *(const v4u*)(fBase + d0);
      b0.u[1] = *(const v4u*)(fBase + d0 + 16);
      b1.u[0] = *(const v4u*)(fBase + 8192 + d0);       // row-subtile 1 (+16 rows)
      b1.u[1] = *(const v4u*)(fBase + 8192 + d0 + 16);
      c00 = __builtin_amdgcn_wmma_f32_16x16x32_f16(false, a0.h, false, b0.h, (short)0, c00, false, false);
      c01 = __builtin_amdgcn_wmma_f32_16x16x32_f16(false, a0.h, false, b1.h, (short)0, c01, false, false);
      c10 = __builtin_amdgcn_wmma_f32_16x16x32_f16(false, a1.h, false, b0.h, (short)0, c10, false, false);
      c11 = __builtin_amdgcn_wmma_f32_16x16x32_f16(false, a1.h, false, b1.h, (short)0, c11, false, false);
    }

    float t;
    // e-subtile 0 contributions
    t = c00[0] + be0a.x; accp0 += t * t;   t = c01[0] + be0a.x; accp1 += t * t;
    t = c00[1] + be0a.y; accp0 += t * t;   t = c01[1] + be0a.y; accp1 += t * t;
    t = c00[2] + be0a.z; accp0 += t * t;   t = c01[2] + be0a.z; accp1 += t * t;
    t = c00[3] + be0a.w; accp0 += t * t;   t = c01[3] + be0a.w; accp1 += t * t;
    t = c00[4] + be0b.x; accp0 += t * t;   t = c01[4] + be0b.x; accp1 += t * t;
    t = c00[5] + be0b.y; accp0 += t * t;   t = c01[5] + be0b.y; accp1 += t * t;
    t = c00[6] + be0b.z; accp0 += t * t;   t = c01[6] + be0b.z; accp1 += t * t;
    t = c00[7] + be0b.w; accp0 += t * t;   t = c01[7] + be0b.w; accp1 += t * t;
    // e-subtile 1 contributions
    t = c10[0] + be1a.x; accp0 += t * t;   t = c11[0] + be1a.x; accp1 += t * t;
    t = c10[1] + be1a.y; accp0 += t * t;   t = c11[1] + be1a.y; accp1 += t * t;
    t = c10[2] + be1a.z; accp0 += t * t;   t = c11[2] + be1a.z; accp1 += t * t;
    t = c10[3] + be1a.w; accp0 += t * t;   t = c11[3] + be1a.w; accp1 += t * t;
    t = c10[4] + be1b.x; accp0 += t * t;   t = c11[4] + be1b.x; accp1 += t * t;
    t = c10[5] + be1b.y; accp0 += t * t;   t = c11[5] + be1b.y; accp1 += t * t;
    t = c10[6] + be1b.z; accp0 += t * t;   t = c11[6] + be1b.z; accp1 += t * t;
    t = c10[7] + be1b.w; accp0 += t * t;   t = c11[7] + be1b.w; accp1 += t * t;

    if (wave == 0) wait_tensor0();  // staged tile landed (overlapped above)
    __syncthreads();                // all waves done reading current buffer
  }

  // Merge the two lane halves (each covered half of this wave's e range).
  accp0 += __shfl_xor(accp0, 16, 32);
  accp1 += __shfl_xor(accp1, 16, 32);

  // Cross-wave reduction over the 2 e-groups via LDS (reuse feature tile area).
  float* red = (float*)lds;   // [2][128]
  if (lane < 16) {
    red[eGrp * MTILE + rowGrp * 32 + lane]      = accp0;
    red[eGrp * MTILE + rowGrp * 32 + 16 + lane] = accp1;
  }
  __syncthreads();
  if (tid < MTILE) {
    float s = red[tid] + red[MTILE + tid];
    out[(size_t)(m0 + tid) * KCLS + k] = sqrtf(s);
  }
}

// ---------------------------------------------------------------------------
extern "C" void kernel_launch(void* const* d_in, const int* in_sizes, int n_in,
                              void* d_out, int out_size, void* d_ws, size_t ws_size,
                              hipStream_t stream) {
  const float* features = (const float*)d_in[0];  // [4096][512]
  const float* A        = (const float*)d_in[1];  // [100][512][512]
  const float* bias     = (const float*)d_in[2];  // [100][512]
  float* out = (float*)d_out;                     // [4096][100]

  // Workspace: f16 copies of A (52.4 MB) then features (4.2 MB).
  _Float16* aH = (_Float16*)d_ws;
  _Float16* fH = aH + (size_t)KCLS * DDIM * DDIM;

  {
    int n8 = KCLS * DDIM * DDIM / 8;
    cvt_f32_to_f16<<<(n8 + 255) / 256, 256, 0, stream>>>(A, aH, n8);
  }
  {
    int n8 = B_ROWS * DDIM / 8;
    cvt_f32_to_f16<<<(n8 + 255) / 256, 256, 0, stream>>>(features, fH, n8);
  }

  (void)hipFuncSetAttribute(reinterpret_cast<const void*>(rbf_wmma_kernel),
                            hipFuncAttributeMaxDynamicSharedMemorySize, LDS_TOTAL);

  dim3 grid(B_ROWS / MTILE, KCLS);
  rbf_wmma_kernel<<<grid, dim3(256), LDS_TOTAL, stream>>>(fH, aH, bias, out);
}
